// Decoder_17506286698673
// MI455X (gfx1250) — matrix operand order repaired and checked by reading, not compiled, not run on hardware
//
#include <hip/hip_runtime.h>
#include <hip/hip_bf16.h>
#include <float.h>

// Problem sizes (match reference)
constexpr int VSZ = 32000;
constexpr int ESZ = 512;
constexpr int HD  = 512;
constexpr int ENC = 512;
constexpr int ASZ = 256;
constexpr int BB  = 16;
constexpr int LL  = 64;
constexpr int SS  = 256;
constexpr int XSZ = ESZ + ENC;     // 1024, GRU input width
constexpr int OSZ = HD + ENC;      // 1024, output-projection K
constexpr int NTOK = BB * LL;      // 1024 tokens

// Workspace layout (bytes)
constexpr size_t OFF_HWH = 0;                                   // [B,S,A] f32 : 4 MiB
constexpr size_t OFF_GIE = OFF_HWH + (size_t)BB*SS*ASZ*4;       // [tok,1536] f32 : 6 MiB
constexpr size_t OFF_CTX = OFF_GIE + (size_t)NTOK*3*HD*4;       // [B,ENC] f32
constexpr size_t OFF_H   = OFF_CTX + (size_t)BB*ENC*4;          // 2 x [B,HD] f32 ping-pong
constexpr size_t OFF_OBF = OFF_H   + (size_t)2*BB*HD*4;         // [tok,1024] bf16 : 2 MiB
constexpr size_t OFF_WBF = OFF_OBF + (size_t)NTOK*OSZ*2;        // [V,1024] bf16 : 62.5 MiB

typedef __attribute__((ext_vector_type(16))) __bf16 v16bf;
typedef __attribute__((ext_vector_type(8)))  float  v8f;

__device__ __forceinline__ unsigned short f2bf(float f) {
    unsigned u = __float_as_uint(f);
    unsigned r = u + 0x7FFFu + ((u >> 16) & 1u);   // round-to-nearest-even
    return (unsigned short)(r >> 16);
}

// ---------------- Precompute: GIe[tok, j] = b_ih[j] + emb_W[y[tok]] . W_ih[j, 0:E]
__global__ void k_gie(const int* __restrict__ y, const float* __restrict__ embW,
                      const float* __restrict__ W_ih, const float* __restrict__ b_ih,
                      float* __restrict__ GIe) {
    int idx = blockIdx.x * 256 + threadIdx.x;      // over NTOK*1536
    int tok = idx / (3 * HD);
    int j   = idx % (3 * HD);
    const float* er = embW + (size_t)y[tok] * ESZ;
    const float* wr = W_ih + (size_t)j * XSZ;      // embedding half: cols [0,E)
    float acc = b_ih[j];
    #pragma unroll 4
    for (int e = 0; e < ESZ; ++e) acc += er[e] * wr[e];
    GIe[idx] = acc;
}

// ---------------- Precompute: HWh[b,s,a] = H_sent[b,s,:] . attn_Wh[a,:]
__global__ void k_hwh(const float* __restrict__ H, const float* __restrict__ Wh,
                      float* __restrict__ HWh) {
    int idx = blockIdx.x * 256 + threadIdx.x;      // over B*S*A
    int a  = idx % ASZ;
    int bs = idx / ASZ;
    const float* hr = H  + (size_t)bs * ENC;
    const float* wr = Wh + (size_t)a  * ENC;
    float acc = 0.f;
    #pragma unroll 4
    for (int e = 0; e < ENC; ++e) acc += hr[e] * wr[e];
    HWh[idx] = acc;
}

// ---------------- fp32 -> bf16 weight conversion (grid-stride)
__global__ void k_cvt(const float* __restrict__ w, unsigned short* __restrict__ o, int n) {
    for (int i = blockIdx.x * blockDim.x + threadIdx.x; i < n; i += gridDim.x * blockDim.x)
        o[i] = f2bf(w[i]);
}

// ---------------- Per-step attention: one block per batch row (256 threads)
__global__ void k_att(int l, const float* __restrict__ h_prev, const float* __restrict__ H,
                      const unsigned char* __restrict__ mask, const float* __restrict__ HWh,
                      const float* __restrict__ Ws, const float* __restrict__ bs,
                      const float* __restrict__ av, float* __restrict__ ctx,
                      unsigned short* __restrict__ Obf) {
    __shared__ float sWs[ASZ];
    __shared__ float prob[SS];
    __shared__ float red[256];
    __shared__ float hsh[HD];
    __shared__ float smax, ssum;
    const int b = blockIdx.x, t = threadIdx.x;

    hsh[t]       = h_prev[b * HD + t];
    hsh[t + 256] = h_prev[b * HD + 256 + t];
    __syncthreads();

    {   // sWs[a] for a == t  (ASZ == blockDim)
        const float* w = Ws + (size_t)t * HD;
        float acc = bs[t];
        #pragma unroll 4
        for (int k = 0; k < HD; ++k) acc += hsh[k] * w[k];
        sWs[t] = acc;
    }
    __syncthreads();

    // score e[s] for s == t
    float e;
    {
        const float* hw = HWh + ((size_t)b * SS + t) * ASZ;
        float acc = 0.f;
        #pragma unroll 4
        for (int a = 0; a < ASZ; ++a) acc += av[a] * tanhf(hw[a] + sWs[a]);
        e = mask[b * SS + t] ? acc : -FLT_MAX;
    }
    // masked softmax over S
    red[t] = e; __syncthreads();
    for (int o = 128; o > 0; o >>= 1) { if (t < o) red[t] = fmaxf(red[t], red[t + o]); __syncthreads(); }
    if (t == 0) smax = red[0];
    __syncthreads();
    float ex = __expf(e - smax);
    red[t] = ex; __syncthreads();
    for (int o = 128; o > 0; o >>= 1) { if (t < o) red[t] += red[t + o]; __syncthreads(); }
    if (t == 0) ssum = red[0];
    __syncthreads();
    prob[t] = ex / ssum;
    __syncthreads();

    // ctx[b, eo] = sum_s prob[s] * H[b,s,eo]; also write bf16 into O[tok, HD+eo]
    const int tok = b * LL + l;
    for (int eo = t; eo < ENC; eo += 256) {
        const float* hp = H + (size_t)b * SS * ENC + eo;
        float c = 0.f;
        #pragma unroll 4
        for (int s = 0; s < SS; ++s) c += prob[s] * hp[(size_t)s * ENC];
        ctx[b * ENC + eo] = c;
        Obf[(size_t)tok * OSZ + HD + eo] = f2bf(c);
    }
}

// ---------------- Per-step GRU cell: thread -> (b,u); blocks of 256, 2 blocks per b
__global__ void k_gru(int l, const float* __restrict__ h_prev, float* __restrict__ h_next,
                      const float* __restrict__ ctx, const float* __restrict__ GIe,
                      const float* __restrict__ W_ih, const float* __restrict__ W_hh,
                      const float* __restrict__ b_hh, unsigned short* __restrict__ Obf) {
    __shared__ float csh[ENC];
    __shared__ float hsh[HD];
    const int b = blockIdx.x >> 1;
    const int u = ((blockIdx.x & 1) * 256) + threadIdx.x;
    const int t = threadIdx.x;
    csh[t]       = ctx[b * ENC + t];
    csh[t + 256] = ctx[b * ENC + 256 + t];
    hsh[t]       = h_prev[b * HD + t];
    hsh[t + 256] = h_prev[b * HD + 256 + t];
    __syncthreads();

    const int tok = b * LL + l;
    float gi[3], gh[3];
    #pragma unroll
    for (int g = 0; g < 3; ++g) {
        const int row = g * HD + u;
        const float* wi = W_ih + (size_t)row * XSZ + ESZ;   // ctx half of W_ih
        const float* wh = W_hh + (size_t)row * HD;
        float a1 = GIe[(size_t)tok * (3 * HD) + row];       // includes b_ih + emb part
        float a2 = b_hh[row];
        #pragma unroll 4
        for (int k = 0; k < HD; ++k) { a1 += csh[k] * wi[k]; a2 += hsh[k] * wh[k]; }
        gi[g] = a1; gh[g] = a2;
    }
    float r  = 1.f / (1.f + __expf(-(gi[0] + gh[0])));
    float z  = 1.f / (1.f + __expf(-(gi[1] + gh[1])));
    float n  = tanhf(gi[2] + r * gh[2]);
    float hn = (1.f - z) * n + z * hsh[u];
    h_next[b * HD + u] = hn;
    Obf[(size_t)tok * OSZ + u] = f2bf(hn);
}

// ---------------- Batched output projection: logits = O @ out_W.T + b  (WMMA bf16)
// One wave computes a 16(M) x 64(N) strip: A fragment reused over 4 B fragments.
__global__ void k_logits(const unsigned short* __restrict__ Obf,
                         const unsigned short* __restrict__ Wbf,
                         const float* __restrict__ outb, float* __restrict__ out) {
    const int wave = (blockIdx.x * 256 + threadIdx.x) >> 5;  // 0..31999
    const int lane = threadIdx.x & 31;
    const int mt = wave / 500;            // 64 M-tiles
    const int ns = wave % 500;            // 500 N-strips of 64
    const int m0 = mt * 16, n0 = ns * 64;
    const int hi = lane >> 4, lp = lane & 15;

    const unsigned short* arow = Obf + (size_t)(m0 + lp) * OSZ;
    v8f acc0 = {}, acc1 = {}, acc2 = {}, acc3 = {};

    for (int k0 = 0; k0 < OSZ; k0 += 32) {
        union { v16bf v; uint4 q[2]; } Af, Bf0, Bf1, Bf2, Bf3;
        // A 16x32 bf16 fragment: lanes<16 hold K {0..15}, lanes>=16 hold K {16..31}
        const unsigned short* ab = arow + k0 + hi * 16;
        Af.q[0] = *(const uint4*)ab;
        Af.q[1] = *(const uint4*)(ab + 8);
        // B 32x16 bf16 fragments: lane = N, 16 contiguous K per lane (hi half-wave -> K+16)
        const unsigned short* bb = Wbf + (size_t)(n0 + lp) * OSZ + k0 + hi * 16;
        Bf0.q[0] = *(const uint4*)bb;                       Bf0.q[1] = *(const uint4*)(bb + 8);
        Bf1.q[0] = *(const uint4*)(bb + 16 * OSZ);          Bf1.q[1] = *(const uint4*)(bb + 16 * OSZ + 8);
        Bf2.q[0] = *(const uint4*)(bb + 32 * OSZ);          Bf2.q[1] = *(const uint4*)(bb + 32 * OSZ + 8);
        Bf3.q[0] = *(const uint4*)(bb + 48 * OSZ);          Bf3.q[1] = *(const uint4*)(bb + 48 * OSZ + 8);
        if (k0 + 32 < OSZ) __builtin_prefetch(bb + 32, 0, 1);   // next K-tile -> global_prefetch

        acc0 = __builtin_amdgcn_wmma_f32_16x16x32_bf16(false, Af.v, false, Bf0.v, (short)0, acc0, false, false);
        acc1 = __builtin_amdgcn_wmma_f32_16x16x32_bf16(false, Af.v, false, Bf1.v, (short)0, acc1, false, false);
        acc2 = __builtin_amdgcn_wmma_f32_16x16x32_bf16(false, Af.v, false, Bf2.v, (short)0, acc2, false, false);
        acc3 = __builtin_amdgcn_wmma_f32_16x16x32_bf16(false, Af.v, false, Bf3.v, (short)0, acc3, false, false);
    }

    // D layout: VGPR r -> M = m0 + r + (hi?8:0); lane lp -> N
    const float b0 = outb[n0 + lp], b1 = outb[n0 + 16 + lp];
    const float b2 = outb[n0 + 32 + lp], b3 = outb[n0 + 48 + lp];
    #pragma unroll
    for (int r = 0; r < 8; ++r) {
        float* orow = out + (size_t)(m0 + r + hi * 8) * VSZ;
        orow[n0 + lp]      = acc0[r] + b0;
        orow[n0 + 16 + lp] = acc1[r] + b1;
        orow[n0 + 32 + lp] = acc2[r] + b2;
        orow[n0 + 48 + lp] = acc3[r] + b3;
    }
}

extern "C" void kernel_launch(void* const* d_in, const int* in_sizes, int n_in,
                              void* d_out, int out_size, void* d_ws, size_t ws_size,
                              hipStream_t stream) {
    const int*           y_in  = (const int*)d_in[0];
    const float*         Hs    = (const float*)d_in[1];
    const unsigned char* mask  = (const unsigned char*)d_in[2];
    const float*         inith = (const float*)d_in[3];
    const float*         embW  = (const float*)d_in[4];
    const float*         W_ih  = (const float*)d_in[5];
    const float*         b_ih  = (const float*)d_in[6];
    const float*         W_hh  = (const float*)d_in[7];
    const float*         b_hh  = (const float*)d_in[8];
    const float*         aWh   = (const float*)d_in[9];
    const float*         aWs   = (const float*)d_in[10];
    const float*         abs_  = (const float*)d_in[11];
    const float*         av    = (const float*)d_in[12];
    const float*         outW  = (const float*)d_in[13];
    const float*         outb  = (const float*)d_in[14];

    char* ws = (char*)d_ws;
    float*          HWh = (float*)(ws + OFF_HWH);
    float*          GIe = (float*)(ws + OFF_GIE);
    float*          ctx = (float*)(ws + OFF_CTX);
    float*          hb  = (float*)(ws + OFF_H);          // two ping-pong halves
    unsigned short* Obf = (unsigned short*)(ws + OFF_OBF);
    unsigned short* Wbf = (unsigned short*)(ws + OFF_WBF);

    // Precompute (parallel, before the scan)
    k_gie<<<(NTOK * 3 * HD) / 256, 256, 0, stream>>>(y_in, embW, W_ih, b_ih, GIe);
    k_hwh<<<(BB * SS * ASZ) / 256, 256, 0, stream>>>(Hs, aWh, HWh);
    k_cvt<<<4096, 256, 0, stream>>>(outW, Wbf, VSZ * OSZ);
    hipMemcpyAsync(hb, inith, (size_t)BB * HD * sizeof(float),
                   hipMemcpyDeviceToDevice, stream);

    // Sequential recurrence (small kernels; logits hoisted out)
    for (int l = 0; l < LL; ++l) {
        const float* hp = hb + (size_t)(l & 1) * BB * HD;
        float*       hn = hb + (size_t)((l + 1) & 1) * BB * HD;
        k_att<<<BB, 256, 0, stream>>>(l, hp, Hs, mask, HWh, aWs, abs_, av, ctx, Obf);
        k_gru<<<2 * BB, 256, 0, stream>>>(l, hp, hn, ctx, GIe, W_ih, W_hh, b_hh, Obf);
    }

    // One big WMMA GEMM: [1024 x 1024] x [1024 x 32000] + bias
    k_logits<<<4000, 256, 0, stream>>>(Obf, Wbf, outb, (float*)d_out);
}
